// ConvSA_70858370449446
// MI455X (gfx1250) — matrix operand order repaired and checked by reading, not compiled, not run on hardware
//
#include <hip/hip_runtime.h>

// ---------------- CDNA5 / gfx1250 ConvSA (bf16 WMMA, async-LDS V staging) ----
// B=8, L=1024, C=512, KH=4, HID=64, NL=2, KS=3

typedef __attribute__((ext_vector_type(16))) __bf16 v16bf;
typedef __attribute__((ext_vector_type(8)))  float  v8f;
typedef unsigned short u16;

#define Bc   8
#define Lc   1024
#define Cc   512
#define KHc  4
#define HIDc 64
#define NLc  2
#define KSc  3

union U16frag {
  v16bf v;
  u16   u[16];
  uint4 q[2];
};

__device__ __forceinline__ u16 f2bf(float f) {
  unsigned int x = __float_as_uint(f);
  unsigned int r = x + 0x7fffu + ((x >> 16) & 1u);  // round-to-nearest-even
  return (u16)(r >> 16);
}

__device__ __forceinline__ v8f wmma_bf16(v16bf a, v16bf b, v8f c) {
  return __builtin_amdgcn_wmma_f32_16x16x32_bf16(
      false, a, false, b, (short)0, c, false, false);
}

// ------------------------------------------------------------------
// Elementwise cast fp32 -> bf16
__global__ void cast_bf16_kernel(const float* __restrict__ src,
                                 u16* __restrict__ dst, long n) {
  long i = (long)blockIdx.x * blockDim.x + threadIdx.x;
  if (i < n) dst[i] = f2bf(src[i]);
}

// Fragment-major B packing: element (k,n) of B[K,N] goes to
//   dst[((kt*(N/16) + nt)*32 + lane)*16 + j]
// with kt=k>>5, hi=(k&31)>>4, j=k&15, nt=n>>4, lo=n&15, lane=hi*16+lo.
__device__ __forceinline__ long bfrag_index(int k, int n, int N) {
  int kt = k >> 5, kr = k & 31;
  int hi = kr >> 4, j = kr & 15;
  int nt = n >> 4, lo = n & 15;
  return (((long)kt * (N >> 4) + nt) * 32 + (hi * 16 + lo)) * 16 + j;
}

// torch-Linear W[No,Ki] (fp32) -> bf16 fragment-major B (K=Ki, N=No)
__global__ void pack_T_frag_kernel(const float* __restrict__ W,
                                   u16* __restrict__ dst, int No, int Ki) {
  long i = (long)blockIdx.x * blockDim.x + threadIdx.x;
  if (i >= (long)No * Ki) return;
  int n = (int)(i / Ki), k = (int)(i % Ki);
  dst[bfrag_index(k, n, No)] = f2bf(W[i]);
}

// conv weight W[o,i,t] -> fragment-major B with k = t*C+i, n = o (K=KS*C, N=C)
__global__ void pack_conv_frag_kernel(const float* __restrict__ W,
                                      u16* __restrict__ dst) {
  long i = (long)blockIdx.x * blockDim.x + threadIdx.x;
  if (i >= (long)Cc * Cc * KSc) return;
  int o   = (int)(i / (Cc * KSc));
  int rem = (int)(i % (Cc * KSc));
  int ii  = rem / KSc;
  int t   = rem % KSc;
  dst[bfrag_index(t * Cc + ii, o, Cc)] = f2bf(W[i]);
}

// ------------------------------------------------------------------
// Generic bf16 WMMA GEMM: out = A[M,K] * Bfrag[K,N] (+bias) (+BN+ReLU)
// One wave per 16x16 tile, 4 waves/block. B is fragment-major -> 2 b128/lane.
// CONV=1: A = xbf[B,L,C] via causal im2col, K = KS*C.
// mode 0: +bias -> bf16 row-major.  mode 1: +bias,BN,ReLU -> bf16 row-major.
// mode 3: +bias -> bf16 V-fragment-major pack [bh][kt][nt][lane][j].
template <int CONV>
__global__ __launch_bounds__(128, 1) void gemm16_wmma(
    const u16* __restrict__ A,
    const u16* __restrict__ Bp,
    const float* __restrict__ bias,
    const float* __restrict__ bng,
    const float* __restrict__ bnb,
    const float* __restrict__ bnm,
    const float* __restrict__ bnv,
    u16* __restrict__ outBf,
    int M, int N, int K, int mode) {
  const int lane  = threadIdx.x & 31;
  const int wave  = threadIdx.x >> 5;
  const int mtile = blockIdx.y * 4 + wave;
  const int ntile = blockIdx.x;
  const int row   = mtile * 16 + (lane & 15);
  const int ncol  = ntile * 16 + (lane & 15);
  const int hi    = lane >> 4;

  v8f c = {};
  const long bstep = (long)(N >> 4) * 512;  // elements per 32-K chunk of Bfrag
  const u16* bp = Bp + ((long)ntile * 32 + lane) * 16;
  for (int kk = 0; kk < K; kk += 32, bp += bstep) {
    // ---- A fragment: lane holds row, K = hi*16 + {0..15}
    U16frag a;
    if (CONV) {
      int b_ = row >> 10;
      int l_ = row & (Lc - 1);
      int t  = kk / Cc;                 // constant per 32-chunk
      int ib = (kk % Cc) + hi * 16;
      int sl = l_ + t - (KSc - 1);      // causal left pad
      if (sl < 0) {
        uint4 z = {0u, 0u, 0u, 0u};
        a.q[0] = z; a.q[1] = z;
      } else {
        const u16* ap = A + ((long)(b_ * Lc + sl) * Cc + ib);
        a.q[0] = *(const uint4*)ap;
        a.q[1] = *(const uint4*)(ap + 8);
      }
    } else {
      const u16* ap = A + (long)row * K + kk + hi * 16;
      a.q[0] = *(const uint4*)ap;
      a.q[1] = *(const uint4*)(ap + 8);
    }
    // ---- B fragment: contiguous 32B per lane in fragment-major layout
    U16frag b;
    b.q[0] = *(const uint4*)bp;
    b.q[1] = *(const uint4*)(bp + 8);
    __builtin_prefetch(bp + bstep, 0, 1);  // global_prefetch, OOB dropped

    c = wmma_bf16(a.v, b.v, c);
  }

  // ---- epilogue (C/D: row = hi*8 + r, col = ncol)
  const float bs = bias ? bias[ncol] : 0.0f;
  float scale = 1.0f, shmean = 0.0f, shbeta = 0.0f;
  if (mode == 1) {
    scale  = bng[ncol] / sqrtf(bnv[ncol] + 1e-5f);
    shmean = bnm[ncol];
    shbeta = bnb[ncol];
  }
#pragma unroll
  for (int r = 0; r < 8; ++r) {
    int mg = mtile * 16 + hi * 8 + r;
    float val = c[r] + bs;
    if (mode == 1) val = fmaxf((val - shmean) * scale + shbeta, 0.0f);
    if (mode == 3) {
      // scatter into V-fragment pack: [b*KH+h][kt][nt][lane][j]
      int b_ = mg >> 10, sl = mg & (Lc - 1);
      int kt = sl >> 5, rr = sl & 31;
      int hj = rr >> 4, j = rr & 15;
      int h_ = ncol >> 7, d = ncol & 127;
      int nt = d >> 4, lo = d & 15;
      long idx = ((((long)(b_ * KHc + h_) * (Lc / 32) + kt) * 8 + nt) * 32 +
                  (hj * 16 + lo)) * 16 + j;
      outBf[idx] = f2bf(val);
    } else {
      outBf[(long)mg * N + ncol] = f2bf(val);
    }
  }
}

// ------------------------------------------------------------------
// Fused attention. grid.x = B*KH, grid.y = L/64. Block = 128 (4 waves);
// wave owns 16 q-rows. Pass 1: causal-limited rowsums (unrolled x2 to hide
// WMMA->VALU latency). Pass 2: recompute score tiles, normalize, write
// graphs, transpose through LDS, att@V with WMMA; V chunk (8KB contiguous
// fragment-pack) staged per-block into LDS with
// global_load_async_to_lds_b128 overlapping the score WMMAs.
__global__ __launch_bounds__(128, 1) void attn_wmma_kernel(
    const u16* __restrict__ qmat,   // [B*L,KH*HID]
    const u16* __restrict__ kmat,   // [B*L,KH*HID]
    const u16* __restrict__ vpack,  // frag-major
    const float* __restrict__ prev, // [B*L,C]
    float* __restrict__ graphs,     // [B*KH,L,L]
    float* __restrict__ outF,       // [B*L,C]
    u16* __restrict__ outBf) {      // [B*L,C]
  __shared__ u16 att_lds[4][16][34];  // per-wave 16x32 bf16 transpose buffer
  __shared__ u16 vtile[4096];         // 8KB staged V chunk (shared by block)

  const int lane = threadIdx.x & 31;
  const int wave = threadIdx.x >> 5;
  const int tid  = threadIdx.x;
  const int bh = blockIdx.x;
  const int bb = bh >> 2;
  const int h  = bh & (KHc - 1);
  const int qbase = blockIdx.y * 64 + wave * 16;
  const int lo = lane & 15;
  const int hi = lane >> 4;

  // ---- Q A-fragments, resident for whole kernel
  const int qrowA = qbase + lo;
  const u16* qp = qmat + ((long)(bb * Lc + qrowA) * (KHc * HIDc) + h * HIDc + hi * 16);
  U16frag qa0, qa1;
  qa0.q[0] = *(const uint4*)qp;
  qa0.q[1] = *(const uint4*)(qp + 8);
  qa1.q[0] = *(const uint4*)(qp + 32);
  qa1.q[1] = *(const uint4*)(qp + 40);

  // ============ pass 1: rowsums of relu^2 masked scores (causal-limited) ====
  float rs[8];
#pragma unroll
  for (int r = 0; r < 8; ++r) rs[r] = 0.0f;

  const int stmax = qbase >> 4;  // tiles fully above diagonal contribute 0
#pragma unroll 2
  for (int st = 0; st <= stmax; ++st) {
    const int scol = st * 16 + lo;
    const u16* kp = kmat + ((long)(bb * Lc + scol) * (KHc * HIDc) + h * HIDc + hi * 16);
    U16frag kb0, kb1;
    kb0.q[0] = *(const uint4*)kp;
    kb0.q[1] = *(const uint4*)(kp + 8);
    kb1.q[0] = *(const uint4*)(kp + 32);
    kb1.q[1] = *(const uint4*)(kp + 40);
    v8f s = {};
    s = wmma_bf16(qa0.v, kb0.v, s);
    s = wmma_bf16(qa1.v, kb1.v, s);
#pragma unroll
    for (int r = 0; r < 8; ++r) {
      int qg = qbase + hi * 8 + r;
      int sg = st * 16 + lo;
      float v = fmaxf(s[r], 0.0f);
      if (sg >= qg) v = 0.0f;  // strict lower-triangular mask
      rs[r] += v * v;
    }
  }
#pragma unroll
  for (int r = 0; r < 8; ++r) {
#pragma unroll
    for (int mk = 1; mk < 16; mk <<= 1) rs[r] += __shfl_xor(rs[r], mk, 32);
  }
  float inv[8];
#pragma unroll
  for (int r = 0; r < 8; ++r) inv[r] = 1.0f / (rs[r] + 1e-16f);

  // ============ pass 2: normalize + graphs + att@V =========================
  v8f acc[8] = {};
  const int ktmax_blk = (blockIdx.y * 64 + 48) >> 5;  // uniform across block
  const int ktmax     = qbase >> 5;                   // this wave's causal limit
  for (int kt = 0; kt <= ktmax_blk; ++kt) {
    __syncthreads();  // WAR: prior iteration's LDS reads done everywhere

    // -- kick async DMA of this kt's V chunk (8KB contiguous) into LDS
    {
      const u16* vsrc = vpack + ((long)bh * (Lc / 32) + kt) * 4096;
#pragma unroll
      for (int it = 0; it < 4; ++it) {
        int s16 = tid + 128 * it;  // b128 slot
        unsigned ldsa = (unsigned)(unsigned long long)&vtile[s16 * 8];
        const u16* g = vsrc + s16 * 8;
        asm volatile("global_load_async_to_lds_b128 %0, %1, off"
                     :: "v"(ldsa), "v"(g) : "memory");
      }
    }

    // -- recompute two 16x16 score tiles, normalize, write graphs, stage LDS
    if (kt <= ktmax) {
#pragma unroll
      for (int t2 = 0; t2 < 2; ++t2) {
        const int st = kt * 2 + t2;
        const int scol = st * 16 + lo;
        const u16* kp = kmat + ((long)(bb * Lc + scol) * (KHc * HIDc) + h * HIDc + hi * 16);
        U16frag kb0, kb1;
        kb0.q[0] = *(const uint4*)kp;
        kb0.q[1] = *(const uint4*)(kp + 8);
        kb1.q[0] = *(const uint4*)(kp + 32);
        kb1.q[1] = *(const uint4*)(kp + 40);
        v8f s = {};
        s = wmma_bf16(qa0.v, kb0.v, s);
        s = wmma_bf16(qa1.v, kb1.v, s);
#pragma unroll
        for (int r = 0; r < 8; ++r) {
          int ml = hi * 8 + r;
          int qg = qbase + ml;
          int sg = st * 16 + lo;
          float v = fmaxf(s[r], 0.0f);
          if (sg >= qg) v = 0.0f;
          float att = v * v * inv[r];
          graphs[((long)bh * Lc + qg) * Lc + sg] = att;
          att_lds[wave][ml][t2 * 16 + lo] = f2bf(att);
        }
      }
    }

    asm volatile("s_wait_asynccnt 0x0" ::: "memory");  // V DMA landed
    asm volatile("s_wait_dscnt 0x0" ::: "memory");     // att_lds stores landed
    __syncthreads();

    if (kt <= ktmax) {
      // A fragment of att (row = lo, K = hi*16 + {0..15})
      U16frag aatt;
#pragma unroll
      for (int j = 0; j < 8; ++j) {
        aatt.u[j]     = att_lds[wave][lo][hi * 16 + j];
        aatt.u[8 + j] = att_lds[wave][lo][hi * 16 + 8 + j];
      }
      // V B-fragments straight from staged LDS: 2x ds b128 per fragment
#pragma unroll
      for (int nt = 0; nt < 8; ++nt) {
        const uint4* vp = (const uint4*)&vtile[nt * 512 + lane * 16];
        U16frag vb;
        vb.q[0] = vp[0];
        vb.q[1] = vp[1];
        acc[nt] = wmma_bf16(aatt.v, vb.v, acc[nt]);
      }
    }
  }

  // ---- epilogue first (ends acc liveness): output = prev + cur
#pragma unroll
  for (int nt = 0; nt < 8; ++nt) {
#pragma unroll
    for (int r = 0; r < 8; ++r) {
      int qg = qbase + hi * 8 + r;
      int cg = h * 128 + nt * 16 + lo;
      long idx = (long)(bb * Lc + qg) * Cc + cg;
      float o = prev[idx] + acc[nt][r];
      outF[idx]  = o;
      outBf[idx] = f2bf(o);
    }
  }

  // -- zero-fill the masked-out (upper) region of graphs beyond the diagonal
  {
    const int zstart = (ktmax + 1) * 32;
    const float4 z = {0.0f, 0.0f, 0.0f, 0.0f};
    for (int m = 0; m < 16; ++m) {
      float* gr = graphs + ((long)bh * Lc + (qbase + m)) * Lc;
      for (int s4 = zstart + lane * 4; s4 < Lc; s4 += 128)
        *(float4*)(gr + s4) = z;
    }
  }
}

// ------------------------------------------------------------------
extern "C" void kernel_launch(void* const* d_in, const int* in_sizes, int n_in,
                              void* d_out, int out_size, void* d_ws, size_t ws_size,
                              hipStream_t stream) {
  const float* input   = (const float*)d_in[0];
  const float* ck_w    = (const float*)d_in[1];
  const float* ck_b    = (const float*)d_in[2];
  const float* ck_g    = (const float*)d_in[3];
  const float* ck_beta = (const float*)d_in[4];
  const float* ck_m    = (const float*)d_in[5];
  const float* ck_v    = (const float*)d_in[6];
  const float* cq_w    = (const float*)d_in[7];
  const float* cq_b    = (const float*)d_in[8];
  const float* cq_g    = (const float*)d_in[9];
  const float* cq_beta = (const float*)d_in[10];
  const float* cq_m    = (const float*)d_in[11];
  const float* cq_v    = (const float*)d_in[12];
  const float* qW = (const float*)d_in[13];
  const float* qb = (const float*)d_in[14];
  const float* kW = (const float*)d_in[15];
  const float* kb = (const float*)d_in[16];
  const float* vW = (const float*)d_in[17];
  const float* vb = (const float*)d_in[18];

  const long M   = (long)Bc * Lc;       // 8192
  const long NX  = M * Cc;              // 4,194,304 elements
  const long NQK = M * (KHc * HIDc);    // 2,097,152

  size_t off = 0;
  auto carve = [&](size_t bytes) {
    size_t p = off;
    off = (off + bytes + 255) & ~(size_t)255;
    return (char*)d_ws + p;
  };
  u16* xbf     = (u16*)carve(NX * 2);
  u16* keybf   = (u16*)carve(NX * 2);
  u16* querybf = (u16*)carve(NX * 2);
  u16* wck     = (u16*)carve((size_t)KSc * Cc * Cc * 2);
  u16* wcq     = (u16*)carve((size_t)KSc * Cc * Cc * 2);
  u16* qWp     = (u16*)carve((size_t)NLc * Cc * (KHc * HIDc) * 2);
  u16* kWp     = (u16*)carve((size_t)NLc * Cc * (KHc * HIDc) * 2);
  u16* vWp     = (u16*)carve((size_t)NLc * Cc * Cc * 2);
  u16* qmat    = (u16*)carve(NQK * 2);
  u16* kmat    = (u16*)carve(NQK * 2);
  u16* vpack   = (u16*)carve(NX * 2);   // V in fragment-major pack
  u16* outbf   = (u16*)carve(NX * 2);

  float* graphs_base  = (float*)d_out;
  float* outputs_base = graphs_base + (size_t)NLc * Bc * KHc * Lc * Lc;

  // ---- stage 0: casts / fragment-major weight packing
  cast_bf16_kernel<<<(int)((NX + 255) / 256), 256, 0, stream>>>(input, xbf, NX);
  {
    long n = (long)Cc * Cc * KSc;
    pack_conv_frag_kernel<<<(int)((n + 255) / 256), 256, 0, stream>>>(ck_w, wck);
    pack_conv_frag_kernel<<<(int)((n + 255) / 256), 256, 0, stream>>>(cq_w, wcq);
  }
  for (int i = 0; i < NLc; ++i) {
    long nqk = (long)(KHc * HIDc) * Cc;
    long nv  = (long)Cc * Cc;
    pack_T_frag_kernel<<<(int)((nqk + 255) / 256), 256, 0, stream>>>(
        qW + i * nqk, qWp + i * nqk, KHc * HIDc, Cc);
    pack_T_frag_kernel<<<(int)((nqk + 255) / 256), 256, 0, stream>>>(
        kW + i * nqk, kWp + i * nqk, KHc * HIDc, Cc);
    pack_T_frag_kernel<<<(int)((nv + 255) / 256), 256, 0, stream>>>(
        vW + i * nv, vWp + i * nv, Cc, Cc);
  }

  // ---- stage 1: causal conv + BN + ReLU as im2col WMMA GEMM (K = KS*C)
  {
    dim3 grid(Cc / 16, (int)(M / 64)), blk(128);
    gemm16_wmma<1><<<grid, blk, 0, stream>>>(xbf, wck, ck_b, ck_g, ck_beta,
                                             ck_m, ck_v, keybf,
                                             (int)M, Cc, KSc * Cc, 1);
    gemm16_wmma<1><<<grid, blk, 0, stream>>>(xbf, wcq, cq_b, cq_g, cq_beta,
                                             cq_m, cq_v, querybf,
                                             (int)M, Cc, KSc * Cc, 1);
  }

  // ---- stage 2: per-layer q/k/v linears + fused attention
  for (int i = 0; i < NLc; ++i) {
    const int NQ = KHc * HIDc;  // 256
    {
      dim3 grid(NQ / 16, (int)(M / 64)), blk(128);
      gemm16_wmma<0><<<grid, blk, 0, stream>>>(
          querybf, qWp + (size_t)i * Cc * NQ, qb + i * NQ,
          nullptr, nullptr, nullptr, nullptr, qmat, (int)M, NQ, Cc, 0);
      gemm16_wmma<0><<<grid, blk, 0, stream>>>(
          keybf, kWp + (size_t)i * Cc * NQ, kb + i * NQ,
          nullptr, nullptr, nullptr, nullptr, kmat, (int)M, NQ, Cc, 0);
    }
    {
      const u16* prevbf = (i == 0) ? xbf : outbf;
      dim3 grid(Cc / 16, (int)(M / 64)), blk(128);
      gemm16_wmma<0><<<grid, blk, 0, stream>>>(
          prevbf, vWp + (size_t)i * Cc * Cc, vb + i * Cc,
          nullptr, nullptr, nullptr, nullptr, vpack, (int)M, Cc, Cc, 3);
    }
    {
      const float* prevf = (i == 0) ? input : (outputs_base + (size_t)(i - 1) * M * Cc);
      float* graphs_i = graphs_base + (size_t)i * Bc * KHc * Lc * Lc;
      float* out_i    = outputs_base + (size_t)i * M * Cc;
      dim3 grid(Bc * KHc, Lc / 64), blk(128);
      attn_wmma_kernel<<<grid, blk, 0, stream>>>(qmat, kmat, vpack, prevf,
                                                 graphs_i, out_i, outbf);
    }
  }
}
